// MultiHeadAttention_24000277250604
// MI455X (gfx1250) — matrix operand order repaired and checked by reading, not compiled, not run on hardware
//
#include <hip/hip_runtime.h>
#include <hip/hip_bf16.h>

#define B_     2
#define S_     2048
#define DIN    1024
#define DMODEL 1024
#define H_     16
#define DK_    64

typedef __bf16 bf16_t;
typedef __bf16 v16bf __attribute__((ext_vector_type(16)));
typedef __bf16 v8bf  __attribute__((ext_vector_type(8)));
typedef float  v8f   __attribute__((ext_vector_type(8)));

__device__ __forceinline__ bf16_t f2bf(float f) { return (bf16_t)f; }

__device__ __forceinline__ v8f zero8() {
  v8f z;
#pragma unroll
  for (int i = 0; i < 8; ++i) z[i] = 0.0f;
  return z;
}

__device__ __forceinline__ v8f wmma_bf16(v16bf a, v16bf b, v8f c) {
  // D = A(16x32 bf16) * B(32x16 bf16) + C(16x16 f32)
  return __builtin_amdgcn_wmma_f32_16x16x32_bf16(false, a, false, b, (short)0, c,
                                                 false, false);
}

// A-operand fragment (16x32, M x K) from row-major [M x K] bf16.
// ISA 7.12.2 A layout: lane(r,h): elems 0..7 -> K=h*8..h*8+7, elems 8..15 -> K=16+h*8..
__device__ __forceinline__ v16bf load_fragA_bf16(const bf16_t* __restrict__ base, int ld, int lane) {
  const int r = lane & 15;
  const int h = (lane >> 4) & 1;
  const bf16_t* p0 = base + (size_t)r * ld + h * 8;
  const bf16_t* p1 = p0 + 16;
  v16bf f;
#pragma unroll
  for (int i = 0; i < 8; ++i) f[i] = p0[i];
#pragma unroll
  for (int i = 0; i < 8; ++i) f[i + 8] = p1[i];
  return f;
}

// B-operand fragment (32x16, K x N) loaded from its transposed N x K row-major view.
// B layout matches A: lane(r,h) holds column n=r, elems 0..7 -> K=h*8..h*8+7,
// elems 8..15 -> K=16+h*8..16+h*8+7.
__device__ __forceinline__ v16bf load_fragB_bf16(const bf16_t* __restrict__ base, int ld, int lane) {
  const int r = lane & 15;
  const int h = (lane >> 4) & 1;
  const bf16_t* p = base + (size_t)r * ld + h * 8;
  v16bf f;
#pragma unroll
  for (int i = 0; i < 8; ++i) f[i] = p[i];
#pragma unroll
  for (int i = 0; i < 8; ++i) f[i + 8] = p[16 + i];
  return f;
}

// ---------------------------------------------------------------------------
// One-shot fp32 -> bf16 conversion, 8 elems/thread, vector loads/stores.
// ---------------------------------------------------------------------------
__global__ void __launch_bounds__(256) cvt_bf16_kernel(const float* __restrict__ in,
                                                       bf16_t* __restrict__ out,
                                                       int n) {
  const int idx = (blockIdx.x * 256 + threadIdx.x) * 8;
  if (idx + 8 > n) return;
  const float4 a = *reinterpret_cast<const float4*>(in + idx);
  const float4 b = *reinterpret_cast<const float4*>(in + idx + 4);
  v8bf o;
  o[0] = f2bf(a.x); o[1] = f2bf(a.y); o[2] = f2bf(a.z); o[3] = f2bf(a.w);
  o[4] = f2bf(b.x); o[5] = f2bf(b.y); o[6] = f2bf(b.z); o[7] = f2bf(b.w);
  *reinterpret_cast<v8bf*>(out + idx) = o;
}

// ---------------------------------------------------------------------------
// Projection GEMM: OUT = X @ W^T + bias.  X: [4096 x 1024] bf16, W: [1024 x 1024]
// bf16 (N x K row-major = B^T view).  Block = 8 waves, wave tile 32(M) x 64(N),
// block tile 128 x 128.  MODE 0: bf16 out head-split [B,H,S,DK] (Q/K).
// MODE 1: bf16 out transposed [B,H,DK,S] (V).
// ---------------------------------------------------------------------------
template <int MODE>
__global__ void __launch_bounds__(256) proj_kernel(const bf16_t* __restrict__ X,
                                                   const bf16_t* __restrict__ W,
                                                   const float* __restrict__ bias,
                                                   bf16_t* __restrict__ out) {
  const int lane = threadIdx.x & 31;
  const int wave = threadIdx.x >> 5;
  const int m0 = blockIdx.x * 128 + (wave >> 1) * 32;
  const int n0 = blockIdx.y * 128 + (wave & 1) * 64;
  const int K = DIN;

  v8f acc[2][4];
#pragma unroll
  for (int a = 0; a < 2; ++a)
#pragma unroll
    for (int t = 0; t < 4; ++t) acc[a][t] = zero8();

  for (int k = 0; k < K; k += 32) {
    __builtin_prefetch(W + (size_t)n0 * K + k + 256, 0, 1);
    v16bf a0 = load_fragA_bf16(X + (size_t)m0 * K + k, K, lane);
    v16bf a1 = load_fragA_bf16(X + (size_t)(m0 + 16) * K + k, K, lane);
#pragma unroll
    for (int t = 0; t < 4; ++t) {
      v16bf b = load_fragB_bf16(W + (size_t)(n0 + t * 16) * K + k, K, lane);
      acc[0][t] = wmma_bf16(a0, b, acc[0][t]);
      acc[1][t] = wmma_bf16(a1, b, acc[1][t]);
    }
  }

  const int nl = lane & 15;
  const int hh = (lane >> 4) & 1;
#pragma unroll
  for (int a = 0; a < 2; ++a)
#pragma unroll
    for (int t = 0; t < 4; ++t) {
      const int ncol = n0 + t * 16 + nl;
      const float bv = bias[ncol];
      const int head = ncol >> 6;      // / DK_
      const int d    = ncol & 63;
#pragma unroll
      for (int i = 0; i < 8; ++i) {
        const int mrow = m0 + a * 16 + i + 8 * hh;
        const int bb = mrow >> 11;     // / S_
        const int s  = mrow & 2047;
        const float v = acc[a][t][i] + bv;
        size_t idx;
        if (MODE == 0) idx = ((size_t)(bb * H_ + head) * S_ + s) * DK_ + d;
        else           idx = ((size_t)(bb * H_ + head) * DK_ + d) * S_ + s;
        out[idx] = f2bf(v);
      }
    }
}

// ---------------------------------------------------------------------------
// Flash attention per (b,h).  Block: 4 waves x 16 q-rows = 64 q-rows.
// Streams 64-key blocks: QK^T WMMA -> mask/scale -> online softmax
// (shfl_xor row reduction across 16-lane halves) -> P via LDS -> P*V WMMA.
// ---------------------------------------------------------------------------
__global__ void __launch_bounds__(128) attn_kernel(const bf16_t* __restrict__ Qh,
                                                   const bf16_t* __restrict__ Kh,
                                                   const bf16_t* __restrict__ Vt,
                                                   const int* __restrict__ mask,
                                                   bf16_t* __restrict__ Xout) {
  __shared__ bf16_t pbuf[4][16 * 64];
  const int lane = threadIdx.x & 31;
  const int wave = threadIdx.x >> 5;
  const int bh = blockIdx.y;           // b*H + h
  const int b  = bh >> 4;
  const int hd = bh & 15;
  const int q0 = blockIdx.x * 64 + wave * 16;

  const bf16_t* Qbase = Qh + ((size_t)bh * S_ + q0) * DK_;
  const bf16_t* Kbase = Kh + (size_t)bh * S_ * DK_;
  const bf16_t* Vbase = Vt + (size_t)bh * DK_ * S_;
  const int*    Mbase = mask + ((size_t)b * S_ + q0) * S_;

  const v16bf qf0 = load_fragA_bf16(Qbase, DK_, lane);        // d = 0..31
  const v16bf qf1 = load_fragA_bf16(Qbase + 32, DK_, lane);   // d = 32..63

  const int nl = lane & 15;
  const int hh = (lane >> 4) & 1;

  float row_m[8], row_l[8];
  v8f o[4];
#pragma unroll
  for (int i = 0; i < 8; ++i) { row_m[i] = -3.0e38f; row_l[i] = 0.0f; }
#pragma unroll
  for (int t = 0; t < 4; ++t) o[t] = zero8();

  for (int kb = 0; kb < S_ / 64; ++kb) {
    const int kk0 = kb * 64;

    // ---- scores = Q * K^T (16 x 64 tile per wave); K rows are the N dim ----
    v8f sacc[4];
#pragma unroll
    for (int t = 0; t < 4; ++t) sacc[t] = zero8();
#pragma unroll
    for (int ks = 0; ks < 2; ++ks) {
      const v16bf aq = ks ? qf1 : qf0;
#pragma unroll
      for (int t = 0; t < 4; ++t) {
        v16bf kf = load_fragB_bf16(Kbase + (size_t)(kk0 + t * 16) * DK_ + ks * 32, DK_, lane);
        sacc[t] = wmma_bf16(aq, kf, sacc[t]);
      }
    }

    // ---- scale + mask (mask>0 -> -1000) ----
#pragma unroll
    for (int t = 0; t < 4; ++t)
#pragma unroll
      for (int i = 0; i < 8; ++i) {
        const float s = sacc[t][i] * 0.125f;   // 1/sqrt(DK)
        const int mv = Mbase[(size_t)(i + 8 * hh) * S_ + kk0 + t * 16 + nl];
        sacc[t][i] = (mv > 0) ? -1000.0f : s;
      }

    // ---- online softmax update ----
#pragma unroll
    for (int i = 0; i < 8; ++i) {
      float tm = fmaxf(fmaxf(sacc[0][i], sacc[1][i]), fmaxf(sacc[2][i], sacc[3][i]));
#pragma unroll
      for (int xm = 1; xm < 16; xm <<= 1) tm = fmaxf(tm, __shfl_xor(tm, xm, 32));
      const float m_new = fmaxf(row_m[i], tm);
      const float corr  = __expf(row_m[i] - m_new);
      row_m[i] = m_new;
      float ps = 0.0f;
#pragma unroll
      for (int t = 0; t < 4; ++t) {
        const float p = __expf(sacc[t][i] - m_new);
        sacc[t][i] = p;
        ps += p;
      }
#pragma unroll
      for (int xm = 1; xm < 16; xm <<= 1) ps += __shfl_xor(ps, xm, 32);
      row_l[i] = row_l[i] * corr + ps;
#pragma unroll
      for (int t = 0; t < 4; ++t) o[t][i] *= corr;
    }

    // ---- P: C-layout -> LDS -> A-layout fragments ----
    bf16_t* pl = pbuf[wave];
#pragma unroll
    for (int t = 0; t < 4; ++t)
#pragma unroll
      for (int i = 0; i < 8; ++i)
        pl[(i + 8 * hh) * 64 + t * 16 + nl] = f2bf(sacc[t][i]);
    __syncthreads();
    const v16bf pa0 = load_fragA_bf16(pl, 64, lane);        // keys 0..31 of block
    const v16bf pa1 = load_fragA_bf16(pl + 32, 64, lane);   // keys 32..63

    // ---- o += P * V  (Vt is [DK x S] row-major = B^T view; d rows are N) ----
#pragma unroll
    for (int ks = 0; ks < 2; ++ks) {
      const v16bf ap = ks ? pa1 : pa0;
#pragma unroll
      for (int t = 0; t < 4; ++t) {
        v16bf vf = load_fragB_bf16(Vbase + (size_t)(t * 16) * S_ + kk0 + ks * 32, S_, lane);
        o[t] = wmma_bf16(ap, vf, o[t]);
      }
    }
    __syncthreads();
  }

  // ---- normalize + write merged-head bf16 X ----
#pragma unroll
  for (int t = 0; t < 4; ++t)
#pragma unroll
    for (int i = 0; i < 8; ++i) {
      const int q   = q0 + i + 8 * hh;
      const int col = hd * DK_ + t * 16 + nl;
      Xout[((size_t)b * S_ + q) * DMODEL + col] = f2bf(o[t][i] / row_l[i]);
    }
}

// ---------------------------------------------------------------------------
// Output projection: out = X(bf16) @ Wo^T + bo, fp32 result.
// ---------------------------------------------------------------------------
__global__ void __launch_bounds__(256) out_proj_kernel(const bf16_t* __restrict__ X,
                                                       const bf16_t* __restrict__ W,
                                                       const float* __restrict__ bias,
                                                       float* __restrict__ out) {
  const int lane = threadIdx.x & 31;
  const int wave = threadIdx.x >> 5;
  const int m0 = blockIdx.x * 128 + (wave >> 1) * 32;
  const int n0 = blockIdx.y * 128 + (wave & 1) * 64;
  const int K = DMODEL;

  v8f acc[2][4];
#pragma unroll
  for (int a = 0; a < 2; ++a)
#pragma unroll
    for (int t = 0; t < 4; ++t) acc[a][t] = zero8();

  for (int k = 0; k < K; k += 32) {
    __builtin_prefetch(W + (size_t)n0 * K + k + 256, 0, 1);
    v16bf a0 = load_fragA_bf16(X + (size_t)m0 * K + k, K, lane);
    v16bf a1 = load_fragA_bf16(X + (size_t)(m0 + 16) * K + k, K, lane);
#pragma unroll
    for (int t = 0; t < 4; ++t) {
      v16bf bfr = load_fragB_bf16(W + (size_t)(n0 + t * 16) * K + k, K, lane);
      acc[0][t] = wmma_bf16(a0, bfr, acc[0][t]);
      acc[1][t] = wmma_bf16(a1, bfr, acc[1][t]);
    }
  }

  const int nl = lane & 15;
  const int hh = (lane >> 4) & 1;
#pragma unroll
  for (int a = 0; a < 2; ++a)
#pragma unroll
    for (int t = 0; t < 4; ++t) {
      const int ncol = n0 + t * 16 + nl;
      const float bv = bias[ncol];
#pragma unroll
      for (int i = 0; i < 8; ++i) {
        const int mrow = m0 + a * 16 + i + 8 * hh;
        out[(size_t)mrow * DMODEL + ncol] = acc[a][t][i] + bv;
      }
    }
}

// ---------------------------------------------------------------------------
extern "C" void kernel_launch(void* const* d_in, const int* in_sizes, int n_in,
                              void* d_out, int out_size, void* d_ws, size_t ws_size,
                              hipStream_t stream) {
  const float* query = (const float*)d_in[0];
  const float* key_  = (const float*)d_in[1];
  const float* value = (const float*)d_in[2];
  const int*   mask  = (const int*)d_in[3];
  const float* Wq = (const float*)d_in[4];
  const float* bq = (const float*)d_in[5];
  const float* Wk = (const float*)d_in[6];
  const float* bk = (const float*)d_in[7];
  const float* Wv = (const float*)d_in[8];
  const float* bv = (const float*)d_in[9];
  const float* Wo = (const float*)d_in[10];
  const float* bo = (const float*)d_in[11];
  float* out = (float*)d_out;

  const size_t actElems = (size_t)B_ * S_ * DIN;      // 4M
  const size_t wElems   = (size_t)DMODEL * DIN;       // 1M
  bf16_t* Qh  = (bf16_t*)d_ws;
  bf16_t* Kh  = Qh + actElems;
  bf16_t* Vt  = Kh + actElems;
  bf16_t* Xa  = Vt + actElems;
  bf16_t* Xq  = Xa + actElems;
  bf16_t* Xk  = Xq + actElems;
  bf16_t* Xv  = Xk + actElems;
  bf16_t* Wqb = Xv + actElems;
  bf16_t* Wkb = Wqb + wElems;
  bf16_t* Wvb = Wkb + wElems;
  bf16_t* Wob = Wvb + wElems;   // total 64 MB of bf16 workspace

  // one-shot bf16 conversions (2048 elems / block)
  cvt_bf16_kernel<<<actElems / 2048, 256, 0, stream>>>(query, Xq, (int)actElems);
  cvt_bf16_kernel<<<actElems / 2048, 256, 0, stream>>>(key_,  Xk, (int)actElems);
  cvt_bf16_kernel<<<actElems / 2048, 256, 0, stream>>>(value, Xv, (int)actElems);
  cvt_bf16_kernel<<<wElems / 2048, 256, 0, stream>>>(Wq, Wqb, (int)wElems);
  cvt_bf16_kernel<<<wElems / 2048, 256, 0, stream>>>(Wk, Wkb, (int)wElems);
  cvt_bf16_kernel<<<wElems / 2048, 256, 0, stream>>>(Wv, Wvb, (int)wElems);
  cvt_bf16_kernel<<<wElems / 2048, 256, 0, stream>>>(Wo, Wob, (int)wElems);

  dim3 pgrid(32, 8), pblk(256);
  proj_kernel<0><<<pgrid, pblk, 0, stream>>>(Xq, Wqb, bq, Qh);
  proj_kernel<0><<<pgrid, pblk, 0, stream>>>(Xk, Wkb, bk, Kh);
  proj_kernel<1><<<pgrid, pblk, 0, stream>>>(Xv, Wvb, bv, Vt);
  attn_kernel<<<dim3(S_ / 64, B_ * H_), 128, 0, stream>>>(Qh, Kh, Vt, mask, Xa);
  out_proj_kernel<<<pgrid, pblk, 0, stream>>>(Xa, Wob, bo, out);
}
